// MultiHeadCrossAttention_14886356648347
// MI455X (gfx1250) — matrix operand order repaired and checked by reading, not compiled, not run on hardware
//
#include <hip/hip_runtime.h>

typedef unsigned short u16;
typedef unsigned int   u32;
typedef unsigned long long u64;
typedef __attribute__((ext_vector_type(16))) __bf16 v16bf;
typedef __attribute__((ext_vector_type(8)))  float  v8f;

#define DMODEL 1024
#define NHEAD  16
#define DH     64
#define BATCH  4
#define SLQ    1024
#define SLK    2048

// ---------------------------------------------------------------- helpers

__device__ __forceinline__ u16 f32_bf16(float f) {
  union { float f; u32 u; } x; x.f = f;
  u32 r = 0x7FFFu + ((x.u >> 16) & 1u);   // round-to-nearest-even
  return (u16)((x.u + r) >> 16);
}

union FragAB {              // 16x32 A-frag or 32x16 B-frag (bf16), 8 VGPRs
  v16bf v;
  u16   u[16];
  uint4 q[2];
};

__device__ __forceinline__ v8f wmma_bf16(v16bf a, v16bf b, v8f c) {
  return __builtin_amdgcn_wmma_f32_16x16x32_bf16(
      /*neg_a=*/false, a, /*neg_b=*/false, b,
      /*c_mod=*/(short)0, c, /*reuse_a=*/false, /*reuse_b=*/false);
}

// async 16B/lane copy global -> LDS (tracked by ASYNCcnt)
__device__ __forceinline__ void async_cp16(u32 lds_addr, const void* gptr) {
  asm volatile("global_load_async_to_lds_b128 %0, %1, off"
               :: "v"(lds_addr), "v"((u64)(size_t)gptr) : "memory");
}

// =======================================================================
// Prep kernel 1: f32 -> bf16 (activations), 4 floats / thread
// =======================================================================
__global__ __launch_bounds__(256) void cvt_bf16_kernel(
    const float* __restrict__ src, u16* __restrict__ dst, int n4)
{
  int i = blockIdx.x * 256 + threadIdx.x;
  if (i < n4) {
    float4 f = ((const float4*)src)[i];
    uint2 r;
    r.x = (u32)f32_bf16(f.x) | ((u32)f32_bf16(f.y) << 16);
    r.y = (u32)f32_bf16(f.z) | ((u32)f32_bf16(f.w) << 16);
    ((uint2*)dst)[i] = r;
  }
}

// =======================================================================
// Prep kernel 2: W (f32, [k][n]) -> Wt (bf16, [n][k]) via LDS transpose
// =======================================================================
__global__ __launch_bounds__(256) void wt_kernel(
    const float* __restrict__ W, u16* __restrict__ Wt)
{
  __shared__ float T[32][33];
  int tx = threadIdx.x & 31, ty = threadIdx.x >> 5;
  int n0 = blockIdx.x * 32, k0 = blockIdx.y * 32;
#pragma unroll
  for (int i = 0; i < 4; ++i)
    T[ty + i * 8][tx] = W[(size_t)(k0 + ty + i * 8) * DMODEL + n0 + tx];
  __syncthreads();
#pragma unroll
  for (int i = 0; i < 4; ++i)
    Wt[(size_t)(n0 + ty + i * 8) * DMODEL + k0 + tx] =
        f32_bf16(T[tx][ty + i * 8]);
}

// =======================================================================
// Unified GEMM:  Y = XA(bf16, Mx1024) @ Wt^T(bf16 n-major) + bias
// MODE 0: Y -> q  [b][h][l][d] bf16      MODE 1: Y -> k [b][h][l][d] bf16
// MODE 2: Y -> vT [b][h][d][l] bf16      MODE 3: Y -> f32 row-major (+bias)
// 128x128 tile / WG, 8 waves (2Mx4N), 64x32 per wave, K-step 32,
// double-buffered LDS filled with GLOBAL_LOAD_ASYNC_TO_LDS_B128.
// =======================================================================
template <int MODE, int LSEQ>
__global__ __launch_bounds__(256) void gemm_kernel(
    const u16* __restrict__ XA, const u16* __restrict__ WT,
    const float* __restrict__ bias,
    u16* __restrict__ outb, float* __restrict__ outf)
{
  __shared__ __attribute__((aligned(16))) u16 As[2][128][40]; // [m][k]
  __shared__ __attribute__((aligned(16))) u16 Bs[2][128][40]; // [n][k]

  const int tid  = threadIdx.x;
  const int lane = tid & 31;
  const int wave = tid >> 5;
  const int ll   = lane & 15;
  const int lh   = lane >> 4;
  const int n0   = blockIdx.x * 128;
  const int m0   = blockIdx.y * 128;
  const int wm   = (wave >> 2) * 64;
  const int wn   = (wave & 3) * 32;

  // 512 16B-slots per 128x32 tile; each wave copies 64 A + 64 B slots.
  auto issue_tile = [&](int buf, int kb) {
#pragma unroll
    for (int c = 0; c < 2; ++c) {
      int slot = wave * 64 + c * 32 + lane;
      int row  = slot >> 2;
      int seg  = (slot & 3) * 8;               // u16 offset of 16B segment
      async_cp16((u32)(size_t)&As[buf][row][seg],
                 XA + (size_t)(m0 + row) * DMODEL + kb + seg);
      async_cp16((u32)(size_t)&Bs[buf][row][seg],
                 WT + (size_t)(n0 + row) * DMODEL + kb + seg);
    }
  };

  v8f acc[4][2];
  const v8f vzero = {0.f,0.f,0.f,0.f,0.f,0.f,0.f,0.f};
#pragma unroll
  for (int i = 0; i < 4; ++i)
#pragma unroll
    for (int j = 0; j < 2; ++j) acc[i][j] = vzero;

  issue_tile(0, 0);

  for (int kt = 0; kt < DMODEL / 32; ++kt) {
    const int buf = kt & 1;
    if (kt + 1 < DMODEL / 32) {
      issue_tile(buf ^ 1, (kt + 1) * 32);      // prefetch next tile
      asm volatile("s_wait_asynccnt 0x4" ::: "memory");  // cur tile landed
    } else {
      asm volatile("s_wait_asynccnt 0x0" ::: "memory");
    }
    __syncthreads();                           // all waves' parts visible

    FragAB a[4], b[2];
#pragma unroll
    for (int i = 0; i < 4; ++i) {
      int row = wm + i * 16 + ll;
      int ko  = lh * 8;
      a[i].q[0] = *(const uint4*)&As[buf][row][ko];
      a[i].q[1] = *(const uint4*)&As[buf][row][ko + 16];
    }
#pragma unroll
    for (int j = 0; j < 2; ++j) {
      int col = wn + j * 16 + ll;
      int ks  = lh * 8;
      b[j].q[0] = *(const uint4*)&Bs[buf][col][ks];
      b[j].q[1] = *(const uint4*)&Bs[buf][col][ks + 16];
    }
#pragma unroll
    for (int i = 0; i < 4; ++i)
#pragma unroll
      for (int j = 0; j < 2; ++j)
        acc[i][j] = wmma_bf16(a[i].v, b[j].v, acc[i][j]);
    __syncthreads();                           // done reading before refill
  }

  // ---- epilogue
#pragma unroll
  for (int i = 0; i < 4; ++i) {
#pragma unroll
    for (int j = 0; j < 2; ++j) {
      int coln = n0 + wn + j * 16 + ll;
      float bv = bias[coln];
      int h = coln >> 6;
      int d = coln & 63;
#pragma unroll
      for (int r = 0; r < 8; ++r) {
        int row = m0 + wm + i * 16 + r + 8 * lh;
        float val = acc[i][j][r] + bv;
        if (MODE == 3) {
          outf[(size_t)row * DMODEL + coln] = val;
        } else {
          int bb = row / LSEQ;
          int l  = row - bb * LSEQ;
          size_t oi;
          if (MODE == 2) oi = ((size_t)((bb * NHEAD + h) * DH + d)) * LSEQ + l;
          else           oi = ((size_t)(bb * NHEAD + h) * LSEQ + l) * DH + d;
          outb[oi] = f32_bf16(val);
        }
      }
    }
  }
}

// =======================================================================
// Flash attention: one wave (32 threads) per 16 query rows per (b,h).
// q:[b][h][LQ][64] bf16, k:[b][h][LK][64] bf16, vT:[b][h][64][LK] bf16
// ctx out: [b][LQ][DMODEL] bf16 (heads merged)
// =======================================================================
__global__ __launch_bounds__(32) void attn_kernel(
    const u16* __restrict__ qm, const u16* __restrict__ km,
    const u16* __restrict__ vm, const unsigned char* __restrict__ mask,
    u16* __restrict__ ctx)
{
  __shared__ __attribute__((aligned(16))) u16 Ps[16][72]; // 16x64 P tile

  const int lane = threadIdx.x;
  const int ll   = lane & 15;
  const int lh   = lane >> 4;
  const int qt   = blockIdx.x;
  const int h    = blockIdx.y;
  const int b    = blockIdx.z;

  const u16* qp = qm + ((size_t)(b * NHEAD + h) * SLQ + qt * 16) * DH;
  const u16* kp = km + (size_t)(b * NHEAD + h) * SLK * DH;
  const u16* vp = vm + (size_t)(b * NHEAD + h) * DH * SLK;
  const unsigned char* mp = mask + (size_t)b * SLK;

  FragAB qa[2];
#pragma unroll
  for (int s = 0; s < 2; ++s) {
    int ko = s * 32 + lh * 8;
    qa[s].q[0] = *(const uint4*)(qp + (size_t)ll * DH + ko);
    qa[s].q[1] = *(const uint4*)(qp + (size_t)ll * DH + ko + 16);
  }

  const v8f vzero = {0.f,0.f,0.f,0.f,0.f,0.f,0.f,0.f};
  v8f acc[4];
#pragma unroll
  for (int dt = 0; dt < 4; ++dt) acc[dt] = vzero;
  float m8[8], l8[8];
#pragma unroll
  for (int r = 0; r < 8; ++r) { m8[r] = -1.0e30f; l8[r] = 0.f; }

  for (int kb0 = 0; kb0 < SLK; kb0 += 64) {
    // ---------- S = Q K^T for a 16x64 block ----------
    v8f sf[4];
#pragma unroll
    for (int nt = 0; nt < 4; ++nt) {
      int key = kb0 + nt * 16 + ll;
      const u16* kr = kp + (size_t)key * DH;
      FragAB bk0, bk1;
      bk0.q[0] = *(const uint4*)(kr + lh * 8);
      bk0.q[1] = *(const uint4*)(kr + lh * 8 + 16);
      bk1.q[0] = *(const uint4*)(kr + 32 + lh * 8);
      bk1.q[1] = *(const uint4*)(kr + 32 + lh * 8 + 16);
      v8f t = wmma_bf16(qa[0].v, bk0.v, vzero);
      sf[nt] = wmma_bf16(qa[1].v, bk1.v, t);
    }
    // ---------- scale + key-padding mask ----------
#pragma unroll
    for (int nt = 0; nt < 4; ++nt) {
      int key = kb0 + nt * 16 + ll;
      bool mk = mp[key] != 0;
#pragma unroll
      for (int r = 0; r < 8; ++r)
        sf[nt][r] = mk ? -3.0e38f : sf[nt][r] * 0.125f;
    }
    // ---------- online softmax stats ----------
    float bm[8];
#pragma unroll
    for (int r = 0; r < 8; ++r)
      bm[r] = fmaxf(fmaxf(sf[0][r], sf[1][r]), fmaxf(sf[2][r], sf[3][r]));
#pragma unroll
    for (int off = 1; off <= 8; off <<= 1)
#pragma unroll
      for (int r = 0; r < 8; ++r)
        bm[r] = fmaxf(bm[r], __shfl_xor(bm[r], off, 32));

    float mn[8], al[8], rs[8];
#pragma unroll
    for (int r = 0; r < 8; ++r) {
      mn[r] = fmaxf(m8[r], bm[r]);
      al[r] = __expf(m8[r] - mn[r]);
      m8[r] = mn[r];
      rs[r] = 0.f;
    }
#pragma unroll
    for (int nt = 0; nt < 4; ++nt)
#pragma unroll
      for (int r = 0; r < 8; ++r) {
        float p = __expf(sf[nt][r] - mn[r]);
        sf[nt][r] = p;
        rs[r] += p;
      }
#pragma unroll
    for (int off = 1; off <= 8; off <<= 1)
#pragma unroll
      for (int r = 0; r < 8; ++r)
        rs[r] += __shfl_xor(rs[r], off, 32);
#pragma unroll
    for (int r = 0; r < 8; ++r) l8[r] = l8[r] * al[r] + rs[r];
#pragma unroll
    for (int dt = 0; dt < 4; ++dt)
#pragma unroll
      for (int r = 0; r < 8; ++r) acc[dt][r] *= al[r];

    // ---------- transpose P (D-layout f32 -> A-layout bf16) via LDS ----
#pragma unroll
    for (int nt = 0; nt < 4; ++nt)
#pragma unroll
      for (int r = 0; r < 8; ++r)
        Ps[r + 8 * lh][nt * 16 + ll] = f32_bf16(sf[nt][r]);
    asm volatile("s_wait_dscnt 0x0" ::: "memory");

    FragAB pa[2];
#pragma unroll
    for (int s = 0; s < 2; ++s) {
      int ko = s * 32 + lh * 8;
      pa[s].q[0] = *(const uint4*)&Ps[ll][ko];
      pa[s].q[1] = *(const uint4*)&Ps[ll][ko + 16];
    }

    // ---------- acc += P @ V ----------
#pragma unroll
    for (int dt = 0; dt < 4; ++dt) {
      int d = dt * 16 + ll;
      const u16* vr = vp + (size_t)d * SLK + kb0;
      FragAB bv0, bv1;
      bv0.q[0] = *(const uint4*)(vr + lh * 8);
      bv0.q[1] = *(const uint4*)(vr + lh * 8 + 16);
      bv1.q[0] = *(const uint4*)(vr + 32 + lh * 8);
      bv1.q[1] = *(const uint4*)(vr + 32 + lh * 8 + 16);
      acc[dt] = wmma_bf16(pa[0].v, bv0.v, acc[dt]);
      acc[dt] = wmma_bf16(pa[1].v, bv1.v, acc[dt]);
    }
  }

  // ---------- normalize, merge heads ----------
#pragma unroll
  for (int dt = 0; dt < 4; ++dt)
#pragma unroll
    for (int r = 0; r < 8; ++r) {
      float o = acc[dt][r] / fmaxf(l8[r], 1.0e-20f);
      int row = qt * 16 + r + 8 * lh;
      ctx[((size_t)(b * SLQ + row)) * DMODEL + h * DH + dt * 16 + ll] =
          f32_bf16(o);
    }
}

// =======================================================================
// Workspace layout (u16 elements), total 41,943,040 elems = 80 MB:
//   qb   @ 0          [4][16][1024][64]
//   kb   @ 4194304    [4][16][2048][64]
//   vtb  @ 12582912   [4][16][64][2048]
//   ctx  @ 20971520   [4][1024][1024]
//   xq   @ 25165824   [4096][1024]   (q_input bf16)
//   xkv  @ 29360128   [8192][1024]   (kv_input bf16)
//   wq_t @ 37748736   [1024][1024]   (W_Q^T bf16)  +1048576 each for k/v/o
// =======================================================================
extern "C" void kernel_launch(void* const* d_in, const int* in_sizes, int n_in,
                              void* d_out, int out_size, void* d_ws, size_t ws_size,
                              hipStream_t stream) {
  const float* q_input  = (const float*)d_in[0];
  const float* kv_input = (const float*)d_in[1];
  const unsigned char* kmask = (const unsigned char*)d_in[2]; // jax bool = 1B
  const float* W_Q = (const float*)d_in[3];
  const float* b_Q = (const float*)d_in[4];
  const float* W_K = (const float*)d_in[5];
  const float* b_K = (const float*)d_in[6];
  const float* W_V = (const float*)d_in[7];
  const float* b_V = (const float*)d_in[8];
  const float* W_O = (const float*)d_in[9];
  const float* b_O = (const float*)d_in[10];

  u16* ws   = (u16*)d_ws;
  u16* qb   = ws;
  u16* kb   = ws + 4194304;
  u16* vtb  = ws + 12582912;
  u16* ctx  = ws + 20971520;
  u16* xq   = ws + 25165824;
  u16* xkv  = ws + 29360128;
  u16* wq_t = ws + 37748736;
  u16* wk_t = wq_t + 1048576;
  u16* wv_t = wk_t + 1048576;
  u16* wo_t = wv_t + 1048576;

  // prep: bf16 conversions (+ weight transposes)
  cvt_bf16_kernel<<<4096, 256, 0, stream>>>(q_input,  xq,  1048576);
  cvt_bf16_kernel<<<8192, 256, 0, stream>>>(kv_input, xkv, 2097152);
  wt_kernel<<<dim3(32, 32), 256, 0, stream>>>(W_Q, wq_t);
  wt_kernel<<<dim3(32, 32), 256, 0, stream>>>(W_K, wk_t);
  wt_kernel<<<dim3(32, 32), 256, 0, stream>>>(W_V, wv_t);
  wt_kernel<<<dim3(32, 32), 256, 0, stream>>>(W_O, wo_t);

  // projections (async-LDS double-buffered WMMA GEMMs)
  gemm_kernel<0, SLQ><<<dim3(DMODEL / 128, (BATCH * SLQ) / 128), 256, 0, stream>>>(
      xq, wq_t, b_Q, qb, nullptr);
  gemm_kernel<1, SLK><<<dim3(DMODEL / 128, (BATCH * SLK) / 128), 256, 0, stream>>>(
      xkv, wk_t, b_K, kb, nullptr);
  gemm_kernel<2, SLK><<<dim3(DMODEL / 128, (BATCH * SLK) / 128), 256, 0, stream>>>(
      xkv, wv_t, b_V, vtb, nullptr);

  // flash attention
  attn_kernel<<<dim3(SLQ / 16, NHEAD, BATCH), 32, 0, stream>>>(
      qb, kb, vtb, kmask, ctx);

  // output projection -> f32 d_out
  gemm_kernel<3, SLQ><<<dim3(DMODEL / 128, (BATCH * SLQ) / 128), 256, 0, stream>>>(
      ctx, wo_t, b_O, nullptr, (float*)d_out);
}
